// SparseLocalAttention_22127671509477
// MI455X (gfx1250) — compile-verified, matrix operand order checked by reading, not run on hardware
//
#include <hip/hip_runtime.h>

// SparseLocalAttention for MI455X (gfx1250), compile-only target.
// B=1, S=2048, D=1536, H=24, HD=64, WINDOW=512, N_ANCHORS=4.
// f16 WMMA (v_wmma_f32_16x16x32_f16) everywhere, f32 accumulation,
// flash-style streaming softmax, LDS-staged tiles, global_prefetch_b8.
// Workspace: 3 * 24*2048*64 f16 buffers = 18 MB (Q,K rope'd; V plain).

#define S_LEN     2048
#define D_MODEL   1536
#define N_HEADS   24
#define HEAD_DIM  64
#define WIN       512
#define N_ANCH    4

typedef __attribute__((ext_vector_type(16))) _Float16 v16h;
typedef __attribute__((ext_vector_type(8)))  _Float16 v8h;
typedef __attribute__((ext_vector_type(8)))  float    v8f;

// ---------------------------------------------------------------------------
// WMMA helpers
// ---------------------------------------------------------------------------
__device__ __forceinline__ v8f wmma_f16(v16h a, v16h b, v8f c) {
  // D = A(16x32 f16) * B(32x16 f16) + C(16x16 f32)
  return __builtin_amdgcn_wmma_f32_16x16x32_f16(
      /*neg_a=*/false, a, /*neg_b=*/false, b,
      /*c_mod=*/(short)0, c, /*reuse_a=*/false, /*reuse_b=*/false);
}

// Load a 16x32 f16 A-fragment from a row-major [16][K] tile, or a 32x16
// B-fragment from a [n][k] (K-contiguous) tile.
// ISA layout (CDNA5 ISA): lane half h holds K in [8h,8h+8) and [16+8h,16+8h+8).
__device__ __forceinline__ v16h load_frag16(const _Float16* base, int strideHalves, int lane) {
  const int half = (lane >> 4) & 1;
  const int row  = lane & 15;
  const _Float16* p = base + row * strideHalves + half * 8;
  v8h lo = *(const v8h*)(p);        // K = 8h .. 8h+7
  v8h hi = *(const v8h*)(p + 16);   // K = 16+8h .. 16+8h+7
  v16h r;
#pragma unroll
  for (int j = 0; j < 8; ++j) { r[j] = lo[j]; r[j + 8] = hi[j]; }
  return r;
}

// Reductions across the 16 lanes of one half-wave (rows of a C-tile live in a
// 16-lane half; xor masks 1/2/4/8 never cross halves).
__device__ __forceinline__ float halfmax16(float x) {
  x = fmaxf(x, __shfl_xor(x, 1, 32));
  x = fmaxf(x, __shfl_xor(x, 2, 32));
  x = fmaxf(x, __shfl_xor(x, 4, 32));
  x = fmaxf(x, __shfl_xor(x, 8, 32));
  return x;
}
__device__ __forceinline__ float halfsum16(float x) {
  x += __shfl_xor(x, 1, 32);
  x += __shfl_xor(x, 2, 32);
  x += __shfl_xor(x, 4, 32);
  x += __shfl_xor(x, 8, 32);
  return x;
}

// ---------------------------------------------------------------------------
// Kernel 1: QKV projection (y = x @ W^T) + RoPE on Q/K, f16 head-major output.
// Grid: x = 72 column-blocks (3 mats * 24 heads), y = 8 row tiles of 256.
// Block: 256 threads = 8 waves; wave w computes rows [w*32, w*32+32) x 64 cols:
// 2 A-frags reuse 4 B-frags -> 8 WMMA per K-step per wave.
// ---------------------------------------------------------------------------
__global__ __launch_bounds__(256)
void qkv_rope_kernel(const float* __restrict__ x,
                     const float* __restrict__ Wq,
                     const float* __restrict__ Wk,
                     const float* __restrict__ Wv,
                     _Float16* __restrict__ qbuf,
                     _Float16* __restrict__ kbuf,
                     _Float16* __restrict__ vbuf) {
  __shared__ _Float16 aT[256 * 40];  // x tile   [256 rows][32 k], pad to 40
  __shared__ _Float16 bT[64 * 40];   // W tile   [64 cols][32 k],  pad to 40

  const int cb   = blockIdx.x;               // 0..71
  const int rt   = blockIdx.y;               // 0..7
  const int mat  = cb / N_HEADS;             // 0=Q 1=K 2=V
  const int head = cb % N_HEADS;
  const int col0 = head * HEAD_DIM;          // column offset inside this W
  const float* W = (mat == 0) ? Wq : (mat == 1) ? Wk : Wv;

  const int tid  = threadIdx.x;
  const int lane = tid & 31;
  const int wv   = tid >> 5;
  const int s0   = rt * 256;

  v8f acc[2][4] = {};

  for (int kst = 0; kst < D_MODEL; kst += 32) {
    __syncthreads();
    // --- stage A: x[s0+tid][kst..kst+31] -> aT (f32 -> f16), 32 elems/thread
    {
      const float* src = x + (size_t)(s0 + tid) * D_MODEL + kst;
      __builtin_prefetch((const void*)(src + 32), 0, 1);   // next K step
      _Float16* dst = aT + tid * 40;
      _Float16 tmp[32];
#pragma unroll
      for (int j = 0; j < 8; ++j) {
        float4 v = *(const float4*)(src + 4 * j);
        tmp[4 * j + 0] = (_Float16)v.x; tmp[4 * j + 1] = (_Float16)v.y;
        tmp[4 * j + 2] = (_Float16)v.z; tmp[4 * j + 3] = (_Float16)v.w;
      }
#pragma unroll
      for (int j = 0; j < 4; ++j)
        *(v8h*)(dst + 8 * j) = *(const v8h*)(tmp + 8 * j);
    }
    // --- stage B: W[col0+n][kst..], stored [n][k] so B-frags are contiguous
    {
      const int n  = tid >> 2;
      const int kk = (tid & 3) * 8;
      const float* src = W + (size_t)(col0 + n) * D_MODEL + kst + kk;
      __builtin_prefetch((const void*)(src + 32), 0, 1);
      _Float16 tmp[8];
#pragma unroll
      for (int j = 0; j < 2; ++j) {
        float4 v = *(const float4*)(src + 4 * j);
        tmp[4 * j + 0] = (_Float16)v.x; tmp[4 * j + 1] = (_Float16)v.y;
        tmp[4 * j + 2] = (_Float16)v.z; tmp[4 * j + 3] = (_Float16)v.w;
      }
      *(v8h*)(bT + n * 40 + kk) = *(const v8h*)(tmp);
    }
    __syncthreads();

    const v16h a0 = load_frag16(aT + (wv * 32 +  0) * 40, 40, lane);
    const v16h a1 = load_frag16(aT + (wv * 32 + 16) * 40, 40, lane);
#pragma unroll
    for (int c = 0; c < 4; ++c) {
      const v16h b = load_frag16(bT + c * 16 * 40, 40, lane);
      acc[0][c] = wmma_f16(a0, b, acc[0][c]);
      acc[1][c] = wmma_f16(a1, b, acc[1][c]);
    }
  }

  // --- epilogue: RoPE (Q,K) then f16 store head-major [H][S][64]
  const int mrow = ((lane >> 4) & 1) * 8;
  const int ncl  = lane & 15;
  _Float16* outbuf = (mat == 0) ? qbuf : (mat == 1) ? kbuf : vbuf;

#pragma unroll
  for (int r = 0; r < 2; ++r) {
#pragma unroll
    for (int v = 0; v < 8; ++v) {
      const int s = s0 + wv * 32 + r * 16 + mrow + v;
      const size_t rowbase = ((size_t)head * S_LEN + s) * HEAD_DIM;
#pragma unroll
      for (int c = 0; c < 4; ++c) {
        const int n = c * 16 + ncl;            // head-dim index 0..63
        float val = acc[r][c][v];
        if (mat < 2) {
          // partner column n^32 lives in subtile c^2, same lane/vgpr
          const float partner = acc[r][c ^ 2][v];
          const float rot = (n < 32) ? -partner : partner;
          // inv_freq[i] = 10000^(-i/32) = exp(-i * ln(1e4)/32)
          const float inv = __expf(-0.28782313662425572f * (float)(n & 31));
          float sn, cs;
          __sincosf((float)s * inv, &sn, &cs);
          val = val * cs + rot * sn;
        }
        outbuf[rowbase + n] = (_Float16)val;
      }
    }
  }
}

// ---------------------------------------------------------------------------
// Kernel 2: streaming masked attention.
// Grid: x = 16 query tiles of 128, y = 24 heads. Block 256 = 8 waves x 16 q.
// Key blocks of 64 cover [q0-512, q0+128) plus block 0 (anchors); the
// per-element mask (causal & (window | anchor) & spike) resolves overlap.
// ---------------------------------------------------------------------------
__global__ __launch_bounds__(256)
void attn_kernel(const _Float16* __restrict__ qbuf,
                 const _Float16* __restrict__ kbuf,
                 const _Float16* __restrict__ vbuf,
                 const int* __restrict__ spike,
                 float* __restrict__ out) {
  __shared__ _Float16 kT[64 * 72];        // [key][hd]  (B of QK^T)
  __shared__ _Float16 vT[64 * 72];        // [hd][key]  (B of P*V, transposed)
  __shared__ _Float16 pT[8 * 16 * 72];    // per-wave P scratch [16][64]

  const int qt = blockIdx.x, h = blockIdx.y;
  const int q0 = qt * 128;
  const int tid = threadIdx.x, lane = tid & 31, wv = tid >> 5;
  const int mrow = ((lane >> 4) & 1) * 8;
  const int ncl  = lane & 15;

  // Q fragments for this wave's 16 queries, kept in registers for all blocks
  const _Float16* qbase = qbuf + ((size_t)h * S_LEN + q0 + wv * 16) * HEAD_DIM;
  const v16h qa0 = load_frag16(qbase, HEAD_DIM, lane);       // hd  0..31
  const v16h qa1 = load_frag16(qbase + 32, HEAD_DIM, lane);  // hd 32..63

  v8f oacc[4] = {};
  float rmax[8], rsum[8];
#pragma unroll
  for (int v = 0; v < 8; ++v) { rmax[v] = -3e38f; rsum[v] = 0.f; }

  const int b_hi = (q0 + 127) >> 6;
  int b_lo = (q0 - WIN) >> 6; if (b_lo < 0) b_lo = 0;
  const int extra0 = (b_lo > 0) ? 1 : 0;   // prepend anchor block 0 if needed
  const int nblk = extra0 + (b_hi - b_lo + 1);

  for (int bi = 0; bi < nblk; ++bi) {
    const int blk = (extra0 && bi == 0) ? 0 : (b_lo + bi - extra0);
    const int k0  = blk * 64;

    __syncthreads();
    {   // stage K straight, V transposed: 4096 halves each, 16 per thread
      const int key = tid >> 2;
      const int dd  = (tid & 3) * 16;
      const _Float16* ks = kbuf + ((size_t)h * S_LEN + k0 + key) * HEAD_DIM + dd;
#pragma unroll
      for (int j = 0; j < 2; ++j)
        *(v8h*)(kT + key * 72 + dd + 8 * j) = *(const v8h*)(ks + 8 * j);

      const int d   = tid >> 2;
      const int kk0 = (tid & 3) * 16;
      const _Float16* vs = vbuf + ((size_t)h * S_LEN + k0 + kk0) * HEAD_DIM + d;
#pragma unroll
      for (int j = 0; j < 16; ++j)
        vT[d * 72 + kk0 + j] = vs[(size_t)j * HEAD_DIM];
    }
    __syncthreads();

    // --- scores S = Q K^T
    v8f sacc[4] = {};
#pragma unroll
    for (int c = 0; c < 4; ++c) {
      const v16h b0 = load_frag16(kT + c * 16 * 72, 72, lane);
      const v16h b1 = load_frag16(kT + c * 16 * 72 + 32, 72, lane);
      sacc[c] = wmma_f16(qa0, b0, sacc[c]);
      sacc[c] = wmma_f16(qa1, b1, sacc[c]);
    }

    int  kg[4]; bool spk[4];
#pragma unroll
    for (int c = 0; c < 4; ++c) {
      kg[c]  = k0 + c * 16 + ncl;
      spk[c] = spike[kg[c]] != 0;
    }

    // --- online softmax
    float p[4][8];
#pragma unroll
    for (int v = 0; v < 8; ++v) {
      const int q = q0 + wv * 16 + mrow + v;
      float sv[4], bm = -3e38f;
#pragma unroll
      for (int c = 0; c < 4; ++c) {
        const int kk = kg[c];
        const bool m = spk[c] && (kk <= q) && (((q - kk) <= WIN) || (kk < N_ANCH));
        sv[c] = m ? (sacc[c][v] * 0.125f) : -3e38f;   // scale = HD^-0.5
        bm = fmaxf(bm, sv[c]);
      }
      bm = halfmax16(bm);
      const float nm    = fmaxf(rmax[v], bm);
      const float alpha = __expf(rmax[v] - nm);       // 1 when both -inf
      rmax[v] = nm;
      float bs = 0.f;
#pragma unroll
      for (int c = 0; c < 4; ++c) {
        const float pv = (sv[c] > -1e37f) ? __expf(sv[c] - nm) : 0.f;
        p[c][v] = pv; bs += pv;
      }
      bs = halfsum16(bs);
      rsum[v] = rsum[v] * alpha + bs;
#pragma unroll
      for (int c = 0; c < 4; ++c) oacc[c][v] *= alpha;
    }

    // --- P (wave-private LDS) then O += P * V
    _Float16* myP = pT + wv * 16 * 72;
#pragma unroll
    for (int v = 0; v < 8; ++v)
#pragma unroll
      for (int c = 0; c < 4; ++c)
        myP[(mrow + v) * 72 + c * 16 + ncl] = (_Float16)p[c][v];

#pragma unroll
    for (int kh = 0; kh < 2; ++kh) {
      const v16h pa = load_frag16(myP + kh * 32, 72, lane);
#pragma unroll
      for (int c = 0; c < 4; ++c) {
        const v16h vb = load_frag16(vT + c * 16 * 72 + kh * 32, 72, lane);
        oacc[c] = wmma_f16(pa, vb, oacc[c]);
      }
    }
  }

  // --- normalize, zero no-key rows, apply query spike, write f32 output
#pragma unroll
  for (int v = 0; v < 8; ++v) {
    const int q = q0 + wv * 16 + mrow + v;
    const float sc = (rsum[v] > 0.f && spike[q] != 0) ? (1.0f / rsum[v]) : 0.0f;
#pragma unroll
    for (int c = 0; c < 4; ++c)
      out[(size_t)q * D_MODEL + h * HEAD_DIM + c * 16 + ncl] = oacc[c][v] * sc;
  }
}

// ---------------------------------------------------------------------------
extern "C" void kernel_launch(void* const* d_in, const int* in_sizes, int n_in,
                              void* d_out, int out_size, void* d_ws, size_t ws_size,
                              hipStream_t stream) {
  (void)in_sizes; (void)n_in; (void)out_size; (void)ws_size;
  const float* x     = (const float*)d_in[0];
  const int*   spike = (const int*)d_in[1];   // bool mask, assumed int-widened
  const float* Wq    = (const float*)d_in[2];
  const float* Wk    = (const float*)d_in[3];
  const float* Wv    = (const float*)d_in[4];
  float* out = (float*)d_out;

  const size_t hbuf = (size_t)N_HEADS * S_LEN * HEAD_DIM;   // 3,145,728 halves
  _Float16* qbuf = (_Float16*)d_ws;
  _Float16* kbuf = qbuf + hbuf;
  _Float16* vbuf = kbuf + hbuf;                              // total 18 MB

  qkv_rope_kernel<<<dim3(72, 8), 256, 0, stream>>>(x, Wq, Wk, Wv, qbuf, kbuf, vbuf);
  attn_kernel<<<dim3(16, 24), 256, 0, stream>>>(qbuf, kbuf, vbuf, spike, out);
}
